// RecurrentLayer_62362925138231
// MI455X (gfx1250) — compile-verified, matrix operand order checked by reading, not run on hardware
//
#include <hip/hip_runtime.h>

// Problem dims (from reference)
#define T_STEPS 256
#define BATCH   128
#define IN_DIM  512
#define HID     2048
#define ALPHA_C 0.1f

// LDS B-tile: 128 rows x 32 bf16 (64B) + 16B TDM pad per row -> 80B row stride
#define TILE_K       32
#define BTILE_ROWS   128
#define LDS_ROW_ELEM 40   // 32 data + 8 pad (bf16 elements)

typedef __attribute__((ext_vector_type(8)))  __bf16 bf16x8;
typedef __attribute__((ext_vector_type(16))) __bf16 bf16x16;
typedef __attribute__((ext_vector_type(8)))  float  f32x8;
typedef __attribute__((ext_vector_type(4)))  unsigned int v4u;
typedef __attribute__((ext_vector_type(8)))  int v8i;
typedef __attribute__((ext_vector_type(4)))  int v4i;

// ---------------------------------------------------------------------------
// WMMA fragment loaders.
// ISA 16-bit A 16x32 layout: lane L<16 -> row L, halves [0..7]=K 0..7,
// [8..15]=K 16..23; lane>=16 -> row L-16, K shifted by 8. Weights are stored
// (N,K) row-major so the B operand uses the same K-contiguous loader.
// ---------------------------------------------------------------------------
__device__ __forceinline__ bf16x16 load_frag(const __bf16* __restrict__ base,
                                             int stride, int rc_base, int k0,
                                             int lane) {
  const int r = rc_base + (lane & 15);
  const __bf16* p = base + (size_t)r * stride + k0 + ((lane >> 4) << 3);
  bf16x8 lo = *(const bf16x8*)p;
  bf16x8 hi = *(const bf16x8*)(p + 16);
  return __builtin_shufflevector(lo, hi, 0, 1, 2, 3, 4, 5, 6, 7,
                                 8, 9, 10, 11, 12, 13, 14, 15);
}

// Same fragment pattern, but from the TDM-staged (padded) LDS tile.
__device__ __forceinline__ bf16x16 lds_frag(const __bf16* buf, int r_base,
                                            int lane) {
  const int r = r_base + (lane & 15);
  const __bf16* p = buf + r * LDS_ROW_ELEM + ((lane >> 4) << 3);
  bf16x8 lo = *(const bf16x8*)p;        // ds_load_b128
  bf16x8 hi = *(const bf16x8*)(p + 16); // ds_load_b128
  return __builtin_shufflevector(lo, hi, 0, 1, 2, 3, 4, 5, 6, 7,
                                 8, 9, 10, 11, 12, 13, 14, 15);
}

__device__ __forceinline__ f32x8 wmma_bf16(bf16x16 a, bf16x16 b, f32x8 c) {
  return __builtin_amdgcn_wmma_f32_16x16x32_bf16(false, a, false, b,
                                                 (short)0, c, false, false);
}

// ---------------------------------------------------------------------------
// Tensor Data Mover: DMA one 128(rows) x 32(K) bf16 weight tile into LDS.
// Descriptor per CDNA5 ISA:
//  group0: [1:0]=count=1 | lds_addr | global_addr[56:0] | type=2
//  group1: data_size=1(2B), pad_enable=1, pad_interval=3 (16 DWORDs = one 64B
//          row), pad_amount=3 (4 DWORDs = 16B) -> 80B LDS row stride;
//          tensor_dim0=K, tensor_dim1=rows, tile_dim0=32, tile_dim1=128,
//          tensor_dim0_stride=K.  groups 2/3: zero (2-D tensor).
// Completion tracked with TENSORcnt.
// ---------------------------------------------------------------------------
__device__ __forceinline__ void tdm_load_tile(unsigned lds_off,
                                              const __bf16* gtile,
                                              unsigned tensorK,
                                              unsigned tensorRows) {
  unsigned long long ga = (unsigned long long)(uintptr_t)gtile;
  v4u g0;
  g0[0] = 1u;                                   // count=1 (valid descriptor)
  g0[1] = lds_off;                              // lds_addr (bytes)
  g0[2] = (unsigned)ga;                         // global_addr[31:0]
  g0[3] = (unsigned)(((ga >> 32) & 0x1FFFFFFull) | (2u << 30)); // [56:32]|type=2
  v8i g1;
  g1[0] = (int)((1u << 16) | (1u << 20) | (3u << 22) | (3u << 25));
  g1[1] = (int)(tensorK << 16);                 // tensor_dim0 lo16
  g1[2] = (int)((tensorK >> 16) | (tensorRows << 16));
  g1[3] = (int)((tensorRows >> 16) | ((unsigned)TILE_K << 16)); // tile_dim0
  g1[4] = (int)(unsigned)BTILE_ROWS;            // tile_dim1 (tile_dim2=0)
  g1[5] = (int)tensorK;                         // tensor_dim0_stride lo32
  g1[6] = 0;
  g1[7] = 0;
  v4i gz = {0, 0, 0, 0};
#if __clang_major__ >= 23
  v8i gz8 = {0, 0, 0, 0, 0, 0, 0, 0};
  __builtin_amdgcn_tensor_load_to_lds(g0, g1, gz, gz, gz8, 0);
#else
  __builtin_amdgcn_tensor_load_to_lds(g0, g1, gz, gz, 0);
#endif
}

// ---------------------------------------------------------------------------
// One scan step, fused GEMM (block tile 64M x 128N, 8 waves = 2Mx4N of 32x32):
//   acc[m,n] = sum_k h_in[m,k]*Wh[n,k] + sum_i xt[m,i]*Win[n,i]
//   prev'    = (1-a)*prev + a*(acc + b_h[n] + b_in[n])
//   h'       = relu(prev');  out_t = h' (f32); h_out = bf16(h')
// Weights stream via TDM -> double-buffered LDS; activations via global loads.
// ---------------------------------------------------------------------------
__global__ __launch_bounds__(256)
void rnn_step_kernel(const __bf16* __restrict__ h_in,
                     __bf16* __restrict__ h_out,
                     const __bf16* __restrict__ Wh,
                     const __bf16* __restrict__ Win,
                     const __bf16* __restrict__ xt,
                     const float* __restrict__ b_h,
                     const float* __restrict__ b_in,
                     float* __restrict__ prev_h,
                     float* __restrict__ out_t) {
  __shared__ __bf16 smemB[2][BTILE_ROWS * LDS_ROW_ELEM];

  const int lane = threadIdx.x & 31;
  const int wave = threadIdx.x >> 5;
  const int wm = wave & 1;   // 0..1  (M direction)
  const int wn = wave >> 1;  // 0..3  (N direction)
  const int m_base = blockIdx.x * 64 + wm * 32;
  const int n_blk = blockIdx.y * 128;       // block's N base (weight rows)
  const int nloc = wn * 32;                 // wave's N offset inside B tile

  const unsigned lds_off0 = (unsigned)(uintptr_t)&smemB[0][0];
  const unsigned lds_off1 = (unsigned)(uintptr_t)&smemB[1][0];

  f32x8 acc[2][2] = {};

  // Two phases: hidden recurrence (A=h_in, W=Wh, K=HID) then fused input
  // projection (A=xt, W=Win, K=IN_DIM). Same double-buffered TDM pipeline.
  auto run_phase = [&](const __bf16* __restrict__ Abase, int strideA,
                       const __bf16* __restrict__ Wbase, int tensorK) {
    const int nch = tensorK / TILE_K;
    const __bf16* wrow = Wbase + (size_t)n_blk * tensorK;
    if (wave == 0) {
      tdm_load_tile(lds_off0, wrow + 0, (unsigned)tensorK, (unsigned)HID);
      if (nch > 1)
        tdm_load_tile(lds_off1, wrow + TILE_K, (unsigned)tensorK, (unsigned)HID);
    }
    for (int ic = 0; ic < nch; ++ic) {
      if (wave == 0) {
        // In-order TDM completion: chunk ic done when <= (in_flight-1) remain.
        if (ic + 1 < nch) __builtin_amdgcn_s_wait_tensorcnt(1);
        else              __builtin_amdgcn_s_wait_tensorcnt(0);
      }
      __syncthreads();  // publish LDS tile for chunk ic
      const int k0 = ic * TILE_K;
      const __bf16* btile = &smemB[ic & 1][0];
      bf16x16 a0 = load_frag(Abase, strideA, m_base + 0, k0, lane);
      bf16x16 a1 = load_frag(Abase, strideA, m_base + 16, k0, lane);
      bf16x16 b0 = lds_frag(btile, nloc + 0, lane);
      bf16x16 b1 = lds_frag(btile, nloc + 16, lane);
      acc[0][0] = wmma_bf16(a0, b0, acc[0][0]);
      acc[0][1] = wmma_bf16(a0, b1, acc[0][1]);
      acc[1][0] = wmma_bf16(a1, b0, acc[1][0]);
      acc[1][1] = wmma_bf16(a1, b1, acc[1][1]);
      __syncthreads();  // all waves done reading smemB[ic&1]
      if (wave == 0 && ic + 2 < nch) {
        tdm_load_tile((ic & 1) ? lds_off1 : lds_off0,
                      wrow + (size_t)(ic + 2) * TILE_K,
                      (unsigned)tensorK, (unsigned)HID);
      }
    }
  };

  run_phase(h_in, HID, Wh, HID);
  run_phase(xt, IN_DIM, Win, IN_DIM);

  // Epilogue. C/D layout: lane<16 -> N=lane, VGPR j = row j; lane>=16 -> row j+8.
  #pragma unroll
  for (int mi = 0; mi < 2; ++mi) {
    #pragma unroll
    for (int ni = 0; ni < 2; ++ni) {
      const int n = n_blk + nloc + ni * 16 + (lane & 15);
      const int m0 = m_base + mi * 16 + ((lane >> 4) << 3);
      const float bias = b_h[n] + b_in[n];
      f32x8 a = acc[mi][ni];
      #pragma unroll
      for (int j = 0; j < 8; ++j) {
        const int m = m0 + j;
        const size_t idx = (size_t)m * HID + n;
        float prev = prev_h[idx];
        float pn = (1.0f - ALPHA_C) * prev + ALPHA_C * (a[j] + bias);
        prev_h[idx] = pn;
        float h = fmaxf(pn, 0.0f);
        out_t[idx] = h;
        h_out[idx] = (__bf16)h;
      }
    }
  }
}

__global__ void cvt_bf16_kernel(const float* __restrict__ src,
                                __bf16* __restrict__ dst, long n) {
  long i = blockIdx.x * (long)blockDim.x + threadIdx.x;
  long stride = (long)gridDim.x * blockDim.x;
  for (; i < n; i += stride) dst[i] = (__bf16)src[i];
}

__global__ void init_state_kernel(const float* __restrict__ h0,
                                  float* __restrict__ prev_h,
                                  __bf16* __restrict__ h_bf0) {
  int i = blockIdx.x * blockDim.x + threadIdx.x;  // over B*H
  if (i < BATCH * HID) {
    float v = h0[i % HID];              // prev_h0 = broadcast(h0)
    prev_h[i] = v;
    h_bf0[i] = (__bf16)fmaxf(v, 0.0f);  // curr_h0 = relu(prev_h0)
  }
}

extern "C" void kernel_launch(void* const* d_in, const int* in_sizes, int n_in,
                              void* d_out, int out_size, void* d_ws, size_t ws_size,
                              hipStream_t stream) {
  const float* x    = (const float*)d_in[0];  // (T,B,I)
  const float* W_in = (const float*)d_in[1];  // (H,I)
  const float* b_in = (const float*)d_in[2];  // (H)
  const float* W_h  = (const float*)d_in[3];  // (H,H)
  const float* b_h  = (const float*)d_in[4];  // (H)
  const float* h0   = (const float*)d_in[5];  // (H)
  float* out = (float*)d_out;                 // (T,B,H)

  // Workspace carve-up (~46 MB total)
  char* ws = (char*)d_ws;
  auto aln = [](size_t v) { return (v + 255) & ~(size_t)255; };
  __bf16* Wh_bf  = (__bf16*)ws; ws += aln((size_t)HID * HID * 2);
  __bf16* Win_bf = (__bf16*)ws; ws += aln((size_t)HID * IN_DIM * 2);
  __bf16* x_bf   = (__bf16*)ws; ws += aln((size_t)T_STEPS * BATCH * IN_DIM * 2);
  float*  prev_h = (float*)ws;  ws += aln((size_t)BATCH * HID * 4);
  __bf16* hbuf0  = (__bf16*)ws; ws += aln((size_t)BATCH * HID * 2);
  __bf16* hbuf1  = (__bf16*)ws; ws += aln((size_t)BATCH * HID * 2);

  // One-time conversions: weights become L2-resident bf16 (Wh = 8MB << 192MB L2)
  cvt_bf16_kernel<<<1024, 256, 0, stream>>>(W_h, Wh_bf, (long)HID * HID);
  cvt_bf16_kernel<<<512, 256, 0, stream>>>(W_in, Win_bf, (long)HID * IN_DIM);
  cvt_bf16_kernel<<<4096, 256, 0, stream>>>(x, x_bf, (long)T_STEPS * BATCH * IN_DIM);
  init_state_kernel<<<(BATCH * HID + 255) / 256, 256, 0, stream>>>(h0, prev_h, hbuf0);

  // Serial scan: 256 dependent step-GEMMs, ping-pong bf16 h buffers
  dim3 grid(BATCH / 64, HID / 128);
  for (int t = 0; t < T_STEPS; ++t) {
    const __bf16* h_src = (t & 1) ? hbuf1 : hbuf0;
    __bf16* h_dst       = (t & 1) ? hbuf0 : hbuf1;
    rnn_step_kernel<<<grid, 256, 0, stream>>>(
        h_src, h_dst, Wh_bf, Win_bf,
        x_bf + (size_t)t * BATCH * IN_DIM,
        b_h, b_in, prev_h,
        out + (size_t)t * BATCH * HID);
  }
}
